// Model_69922067579057
// MI455X (gfx1250) — matrix operand order repaired and checked by reading, not compiled, not run on hardware
//
#include <hip/hip_runtime.h>
#include <math.h>

#define B_SZ 8
#define T_SC 2
#define BT 16
#define SEQ 720
#define ENC 862
#define MARKC 4
#define NTOK 866
#define MPAD 896
#define DMODEL 256
#define DFF 512
#define DBLK 720
#define DBLKPAD 768
#define NHEAD 8
#define EHEAD 32
#define LAYERS 2
#define EPSV 1e-5f

typedef __attribute__((ext_vector_type(16))) __bf16 v16bf;
typedef __attribute__((ext_vector_type(8))) float v8f;

union Frag {
  v16bf v;
  unsigned short u[16];
  uint4 q[2];
};

__device__ __forceinline__ unsigned short f2bf(float x) {
  unsigned int u = __float_as_uint(x);
  u += 0x7FFFu + ((u >> 16) & 1u);
  return (unsigned short)(u >> 16);
}

__device__ __forceinline__ float warp_sum(float v) {
#pragma unroll
  for (int off = 16; off > 0; off >>= 1) v += __shfl_xor(v, off, 32);
  return v;
}
__device__ __forceinline__ float warp_max(float v) {
#pragma unroll
  for (int off = 16; off > 0; off >>= 1) v = fmaxf(v, __shfl_xor(v, off, 32));
  return v;
}

// ---------------------------------------------------------------------------
// Generic batched GEMM: C[bt][m][n] = sum_k A[bt][m][k] * W[n][k] + bias[n]
// A: bf16 [BT][MPAD][Kk], W: bf16 [>=ceil64(Nn)][Kk] (rows padded).
// One wave -> 32x64 tile: 8 WMMAs per 32-wide K step on 6 fragments.
// Single A/W base pointer per wave; row-block deltas fold into load offsets.
// act: 0=none, 1=exact GELU. transB16: bf16 output transposed (ld = MPAD).
// ---------------------------------------------------------------------------
__global__ void __launch_bounds__(32, 1)
gemm_bf16_kernel(const unsigned short* __restrict__ A,
                 const unsigned short* __restrict__ W,
                 const float* __restrict__ bias,
                 float* __restrict__ Cf,
                 unsigned short* __restrict__ Cb,
                 int Nn, int Kk, int act, int transB16) {
  const int lane = threadIdx.x;
  const int lh = lane >> 4;
  const int l15 = lane & 15;
  const int n0 = blockIdx.x * 64;
  const int m0 = blockIdx.y * 32;
  const int bt = blockIdx.z;
  A += (size_t)bt * MPAD * Kk;
  const size_t cbase = (size_t)bt * MPAD * Nn;

  v8f acc[2][4] = {};
  const size_t rstep = (size_t)16 * Kk;  // 16-row block stride (elements)
  const unsigned short* arow = A + (size_t)(m0 + l15) * Kk;
  const unsigned short* brow = W + (size_t)(n0 + l15) * Kk;

  for (int k0 = 0; k0 < Kk; k0 += 32) {
    Frag af[2], bfr[4];
#pragma unroll
    for (int i = 0; i < 2; ++i) {
      const uint4* pa = (const uint4*)(arow + i * rstep + k0 + lh * 8);
      af[i].q[0] = pa[0]; af[i].q[1] = pa[2];     // split-K halves per A layout
    }
#pragma unroll
    for (int j = 0; j < 4; ++j) {
      const uint4* pb = (const uint4*)(brow + j * rstep + k0 + lh * 8);
      bfr[j].q[0] = pb[0]; bfr[j].q[1] = pb[2];   // split-K halves per A layout
    }
    if (k0 + 128 < Kk) {                          // stream ahead (global_prefetch_b8)
      __builtin_prefetch(arow + k0 + 128, 0, 1);
      __builtin_prefetch(brow + k0 + 128, 0, 1);
    }
#pragma unroll
    for (int i = 0; i < 2; ++i)
#pragma unroll
      for (int j = 0; j < 4; ++j)
        acc[i][j] = __builtin_amdgcn_wmma_f32_16x16x32_bf16(
            false, af[i].v, false, bfr[j].v, (short)0, acc[i][j], false, false);
  }

#pragma unroll
  for (int i = 0; i < 2; ++i) {
    const int rowb = m0 + i * 16 + lh * 8;
#pragma unroll
    for (int j = 0; j < 4; ++j) {
      const int n = n0 + j * 16 + l15;
      if (n >= Nn) continue;
      const float bsv = bias[n];
#pragma unroll
      for (int r = 0; r < 8; ++r) {
        float v = acc[i][j][r] + bsv;
        if (act == 1) v = 0.5f * v * (1.0f + erff(v * 0.70710678118654752f));
        const int rr = rowb + r;
        if (Cf) Cf[cbase + (size_t)rr * Nn + n] = v;
        if (Cb) {
          const size_t bidx = transB16
              ? ((size_t)bt * Nn * MPAD + (size_t)n * MPAD + rr)
              : (cbase + (size_t)rr * Nn + n);
          Cb[bidx] = f2bf(v);
        }
      }
    }
  }
}

// ---------------------------------------------------------------------------
// Flash attention, one wave per (bt, head, 16-query tile), online softmax.
// QK^T: contraction = E = 32 -> one WMMA per 16-key tile.
// P*V: contraction = 32 keys -> 2 WMMAs per chunk, V read transposed (b128).
// Only 2 KB LDS (C-layout -> A-layout staging) => high occupancy per WGP.
// ---------------------------------------------------------------------------
__global__ void __launch_bounds__(32, 1)
attn_kernel(const unsigned short* __restrict__ Q,
            const unsigned short* __restrict__ K,
            const unsigned short* __restrict__ VT,
            unsigned short* __restrict__ O) {
  __shared__ float pbuf[16 * 32];
  const int lane = threadIdx.x;
  const int lh = lane >> 4, l15 = lane & 15;
  const int rt = blockIdx.x, head = blockIdx.y, bt = blockIdx.z;
  const size_t base = (size_t)bt * MPAD * DMODEL + head * EHEAD;
  const unsigned short* vt = VT + ((size_t)bt * DMODEL + head * EHEAD) * MPAD;

  Frag a;
  {
    const unsigned short* qp = Q + base + (size_t)(rt * 16 + l15) * DMODEL;
    a.q[0] = ((const uint4*)(qp + lh * 8))[0];
    a.q[1] = ((const uint4*)(qp + 16 + lh * 8))[0];
  }
  const float scale = 0.17677669529663687f;  // 1/sqrt(32)
  float runmax[8], runsum[8];
#pragma unroll
  for (int r = 0; r < 8; ++r) { runmax[r] = -3.0e38f; runsum[r] = 0.f; }
  v8f o0 = {}; v8f o1 = {};

  for (int kc = 0; kc < MPAD / 32; ++kc) {
    Frag kb0, kb1;
    const unsigned short* kp0 = K + base + (size_t)(kc * 32 + l15) * DMODEL + lh * 8;
    kb0.q[0] = ((const uint4*)kp0)[0]; kb0.q[1] = ((const uint4*)kp0)[2];
    const unsigned short* kp1 = K + base + (size_t)(kc * 32 + 16 + l15) * DMODEL + lh * 8;
    kb1.q[0] = ((const uint4*)kp1)[0]; kb1.q[1] = ((const uint4*)kp1)[2];
    v8f c0 = {}; v8f c1 = {};
    c0 = __builtin_amdgcn_wmma_f32_16x16x32_bf16(false, a.v, false, kb0.v, (short)0, c0, false, false);
    c1 = __builtin_amdgcn_wmma_f32_16x16x32_bf16(false, a.v, false, kb1.v, (short)0, c1, false, false);

    const int key0 = kc * 32 + l15;
    const int key1 = key0 + 16;
#pragma unroll
    for (int r = 0; r < 8; ++r) {
      const float s0 = (key0 < NTOK) ? c0[r] * scale : -3.0e38f;
      const float s1 = (key1 < NTOK) ? c1[r] * scale : -3.0e38f;
      float mx = fmaxf(s0, s1);
#pragma unroll
      for (int off = 8; off > 0; off >>= 1) mx = fmaxf(mx, __shfl_xor(mx, off, 32));
      const float nm = fmaxf(runmax[r], mx);
      const float fctr = __expf(runmax[r] - nm);
      runmax[r] = nm;
      const float p0 = __expf(s0 - nm);
      const float p1 = __expf(s1 - nm);
      float ls = p0 + p1;
#pragma unroll
      for (int off = 8; off > 0; off >>= 1) ls += __shfl_xor(ls, off, 32);
      runsum[r] = runsum[r] * fctr + ls;
      o0[r] *= fctr;
      o1[r] *= fctr;
      const int m = r + lh * 8;
      pbuf[m * 32 + l15] = p0;
      pbuf[m * 32 + 16 + l15] = p1;
    }
    __syncthreads();
    Frag p, b0, b1;
#pragma unroll
    for (int e = 0; e < 16; ++e) {
      const int kk = e + lh * 16;  // contiguous K index (matches V^T)
      p.u[e] = f2bf(pbuf[l15 * 32 + kk]);
    }
    const uint4* pv0 = (const uint4*)(vt + (size_t)l15 * MPAD + kc * 32 + lh * 16);
    b0.q[0] = pv0[0]; b0.q[1] = pv0[1];
    const uint4* pv1 = (const uint4*)(vt + (size_t)(16 + l15) * MPAD + kc * 32 + lh * 16);
    b1.q[0] = pv1[0]; b1.q[1] = pv1[1];
    o0 = __builtin_amdgcn_wmma_f32_16x16x32_bf16(false, p.v, false, b0.v, (short)0, o0, false, false);
    o1 = __builtin_amdgcn_wmma_f32_16x16x32_bf16(false, p.v, false, b1.v, (short)0, o1, false, false);
    __syncthreads();
  }
#pragma unroll
  for (int r = 0; r < 8; ++r) {
    const float inv = 1.0f / runsum[r];
    const int m = rt * 16 + r + lh * 8;
    unsigned short* op = O + base + (size_t)m * DMODEL;
    op[l15] = f2bf(o0[r] * inv);
    op[16 + l15] = f2bf(o1[r] * inv);
  }
}

// ---------------------------------------------------------------------------
// Support kernels
// ---------------------------------------------------------------------------
__global__ void stats_kernel(const float* __restrict__ x, float* __restrict__ meanO,
                             float* __restrict__ stdO) {
  const int id = blockIdx.x;  // bt*ENC + c
  const int bt = id / ENC, c = id % ENC;
  const int lane = threadIdx.x;
  const float* p = x + (size_t)bt * SEQ * ENC + c;
  float s = 0.f, s2 = 0.f;
  for (int t = lane; t < SEQ; t += 32) { const float v = p[(size_t)t * ENC]; s += v; s2 += v * v; }
  s = warp_sum(s); s2 = warp_sum(s2);
  const float m = s * (1.0f / SEQ);
  const float var = s2 * (1.0f / SEQ) - m * m;
  if (lane == 0) { meanO[id] = m; stdO[id] = sqrtf(var + EPSV); }
}

__global__ void build_xin_kernel(const float* __restrict__ x, const float* __restrict__ xm,
                                 const float* __restrict__ meanI, const float* __restrict__ stdI,
                                 unsigned short* __restrict__ xin) {
  const int n = blockIdx.x;
  const int bt = blockIdx.y;
  const int tid = threadIdx.x;
  unsigned short* dst = xin + ((size_t)bt * MPAD + n) * SEQ;
  if (n < ENC) {
    const float m = meanI[bt * ENC + n];
    const float is = 1.0f / stdI[bt * ENC + n];
    const float* p = x + (size_t)bt * SEQ * ENC + n;
    for (int s = tid; s < SEQ; s += blockDim.x) dst[s] = f2bf((p[(size_t)s * ENC] - m) * is);
  } else if (n < NTOK) {
    const float* p = xm + (size_t)bt * SEQ * MARKC + (n - ENC);
    for (int s = tid; s < SEQ; s += blockDim.x) dst[s] = f2bf(p[(size_t)s * MARKC]);
  } else {
    for (int s = tid; s < SEQ; s += blockDim.x) dst[s] = 0;
  }
}

__global__ void ln_kernel(const float* __restrict__ A, const float* __restrict__ S,
                          const float* __restrict__ g, const float* __restrict__ bb,
                          float* __restrict__ outF, unsigned short* __restrict__ outB) {
  const size_t row = blockIdx.x;
  const int lane = threadIdx.x;
  const float* a = A + row * DMODEL;
  const float* sp = S ? (S + row * DMODEL) : nullptr;
  float vals[8];
  float sum = 0.f;
#pragma unroll
  for (int i = 0; i < 8; ++i) {
    const int idx = lane + i * 32;
    const float v = a[idx] - (sp ? sp[idx] : 0.f);
    vals[i] = v; sum += v;
  }
  sum = warp_sum(sum);
  const float m = sum * (1.0f / DMODEL);
  float vs = 0.f;
#pragma unroll
  for (int i = 0; i < 8; ++i) { const float d = vals[i] - m; vs += d * d; }
  vs = warp_sum(vs);
  const float rstd = rsqrtf(vs * (1.0f / DMODEL) + EPSV);
#pragma unroll
  for (int i = 0; i < 8; ++i) {
    const int idx = lane + i * 32;
    const float o = (vals[i] - m) * rstd * g[idx] + bb[idx];
    outF[row * DMODEL + idx] = o;
    if (outB) outB[row * DMODEL + idx] = f2bf(o);
  }
}

__global__ void cast_kernel(const float* __restrict__ a, unsigned short* __restrict__ o, size_t n) {
  const size_t i = (size_t)blockIdx.x * blockDim.x + threadIdx.x;
  if (i < n) o[i] = f2bf(a[i]);
}

// Cast with row padding: dst has rowsPad rows; rows >= rows are zero-filled.
__global__ void cast_pad_kernel(const float* __restrict__ src, unsigned short* __restrict__ dst,
                                int rows, int rowsPad, int cols) {
  const size_t i = (size_t)blockIdx.x * blockDim.x + threadIdx.x;
  const size_t n = (size_t)rowsPad * cols;
  if (i >= n) return;
  const int r = (int)(i / cols);
  const int c = (int)(i % cols);
  dst[i] = (r < rows) ? f2bf(src[(size_t)r * cols + c]) : (unsigned short)0;
}

__global__ void sub_bf16_kernel(const float* __restrict__ a, const float* __restrict__ b,
                                unsigned short* __restrict__ o, size_t n) {
  const size_t i = (size_t)blockIdx.x * blockDim.x + threadIdx.x;
  if (i < n) o[i] = f2bf(a[i] - b[i]);
}

__global__ void gate_mul_kernel(const float* __restrict__ a, const float* __restrict__ b,
                                float* __restrict__ o, size_t n) {
  const size_t i = (size_t)blockIdx.x * blockDim.x + threadIdx.x;
  if (i < n) { const float s = 1.0f / (1.0f + __expf(-a[i])); o[i] = s * b[i]; }
}

__global__ void concat_bf16_kernel(const float* __restrict__ x, const float* __restrict__ y,
                                   unsigned short* __restrict__ o, size_t n) {
  const size_t i = (size_t)blockIdx.x * blockDim.x + threadIdx.x;
  if (i >= n) return;
  const size_t r = i >> 9;
  const int j = (int)(i & 511);
  const float v = (j < DMODEL) ? x[r * DMODEL + j] : y[r * DMODEL + (j - DMODEL)];
  o[i] = f2bf(v);
}

__global__ void gate_sub_kernel(const float* __restrict__ a, const float* __restrict__ b,
                                float* __restrict__ o, size_t n, int first) {
  const size_t i = (size_t)blockIdx.x * blockDim.x + threadIdx.x;
  if (i >= n) return;
  const float s = 1.0f / (1.0f + __expf(-a[i]));
  const float v = s * b[i];
  o[i] = first ? v : (v - o[i]);
}

__global__ void final_kernel(const float* __restrict__ outF, const float* __restrict__ meanI,
                             const float* __restrict__ stdI, float* __restrict__ out) {
  const size_t i = (size_t)blockIdx.x * blockDim.x + threadIdx.x;
  const size_t total = (size_t)B_SZ * SEQ * ENC;
  if (i >= total) return;
  const int b = (int)(i / ((size_t)SEQ * ENC));
  const int rem = (int)(i % ((size_t)SEQ * ENC));
  const int p = rem / ENC;
  const int c = rem % ENC;
  float acc = 0.f;
#pragma unroll
  for (int t = 0; t < T_SC; ++t) {
    const int bt = b * T_SC + t;
    const float v = outF[((size_t)bt * MPAD + c) * DBLK + p];
    acc += v * stdI[bt * ENC + c] + meanI[bt * ENC + c];
  }
  out[i] = 0.5f * acc;
}

// ---------------------------------------------------------------------------
// Host orchestration
// ---------------------------------------------------------------------------
static inline size_t align_up(size_t x) { return (x + 255) & ~(size_t)255; }

extern "C" void kernel_launch(void* const* d_in, const int* in_sizes, int n_in,
                              void* d_out, int out_size, void* d_ws, size_t ws_size,
                              hipStream_t stream) {
  (void)in_sizes; (void)n_in; (void)out_size; (void)ws_size;
  const float* x      = (const float*)d_in[0];
  const float* xmark  = (const float*)d_in[1];
  const float* embedW = (const float*)d_in[4];
  const float* embedB = (const float*)d_in[5];
  const float* Wq = (const float*)d_in[6];
  const float* Wk = (const float*)d_in[7];
  const float* Wv = (const float*)d_in[8];
  const float* Wo = (const float*)d_in[9];
  const float* W3 = (const float*)d_in[10];
  const float* W4 = (const float*)d_in[11];
  const float* bq = (const float*)d_in[12];
  const float* bk = (const float*)d_in[13];
  const float* bv = (const float*)d_in[14];
  const float* bo = (const float*)d_in[15];
  const float* b3 = (const float*)d_in[16];
  const float* b4 = (const float*)d_in[17];
  const float* W1 = (const float*)d_in[18];
  const float* b1 = (const float*)d_in[19];
  const float* W2 = (const float*)d_in[20];
  const float* b2 = (const float*)d_in[21];
  const float* W5 = (const float*)d_in[22];
  const float* b5 = (const float*)d_in[23];
  const float* W6 = (const float*)d_in[24];
  const float* b6 = (const float*)d_in[25];
  const float* ln1g = (const float*)d_in[26];
  const float* ln1b = (const float*)d_in[27];
  const float* ln2g = (const float*)d_in[28];
  const float* ln2b = (const float*)d_in[29];

  char* wp = (char*)d_ws;
  size_t off = 0;
  auto alloc = [&](size_t bytes) -> void* { void* p = wp + off; off = align_up(off + bytes); return p; };

  float* meanB = (float*)alloc((size_t)BT * ENC * 4);
  float* stdB  = (float*)alloc((size_t)BT * ENC * 4);

  unsigned short* wEmbB = (unsigned short*)alloc((size_t)DMODEL * SEQ * 2);
  unsigned short* wqB = (unsigned short*)alloc((size_t)LAYERS * DMODEL * DMODEL * 2);
  unsigned short* wkB = (unsigned short*)alloc((size_t)LAYERS * DMODEL * DMODEL * 2);
  unsigned short* wvB = (unsigned short*)alloc((size_t)LAYERS * DMODEL * DMODEL * 2);
  unsigned short* woB = (unsigned short*)alloc((size_t)LAYERS * DMODEL * DMODEL * 2);
  unsigned short* w3B = (unsigned short*)alloc((size_t)LAYERS * DMODEL * DMODEL * 2);
  unsigned short* w4B = (unsigned short*)alloc((size_t)LAYERS * DMODEL * DMODEL * 2);
  unsigned short* w1B = (unsigned short*)alloc((size_t)LAYERS * DFF * DMODEL * 2);
  unsigned short* w2B = (unsigned short*)alloc((size_t)LAYERS * DMODEL * DFF * 2);
  unsigned short* w5B = (unsigned short*)alloc((size_t)LAYERS * DBLKPAD * DFF * 2);  // row-padded
  unsigned short* w6B = (unsigned short*)alloc((size_t)LAYERS * DBLKPAD * DFF * 2);  // row-padded

  const size_t nTok = (size_t)BT * MPAD;
  unsigned short* xinB = (unsigned short*)alloc(nTok * SEQ * 2);
  float*          hF   = (float*)alloc(nTok * DMODEL * 4);
  unsigned short* hB   = (unsigned short*)alloc(nTok * DMODEL * 2);
  unsigned short* qB   = (unsigned short*)alloc(nTok * DMODEL * 2);
  unsigned short* kB   = (unsigned short*)alloc(nTok * DMODEL * 2);
  unsigned short* vtB  = (unsigned short*)alloc((size_t)BT * DMODEL * MPAD * 2);  // V^T
  unsigned short* attB = (unsigned short*)alloc(nTok * DMODEL * 2);
  float*          xattF = (float*)alloc(nTok * DMODEL * 4);
  float*          xlnF  = (float*)alloc(nTok * DMODEL * 4);
  unsigned short* xlnB  = (unsigned short*)alloc(nTok * DMODEL * 2);
  unsigned short* y1B   = (unsigned short*)alloc(nTok * DFF * 2);
  float*          yF    = (float*)alloc(nTok * DMODEL * 4);
  unsigned short* xsB   = (unsigned short*)alloc(nTok * DMODEL * 2);
  float*          g3F   = (float*)alloc(nTok * DMODEL * 4);
  float*          g4F   = (float*)alloc(nTok * DMODEL * 4);
  float*          hhF   = (float*)alloc(nTok * DMODEL * 4);
  unsigned short* catB  = (unsigned short*)alloc(nTok * DFF * 2);
  float*          g5F   = (float*)alloc(nTok * DBLK * 4);
  float*          g6F   = (float*)alloc(nTok * DBLK * 4);
  float*          outF  = (float*)alloc(nTok * DBLK * 4);

  auto cast = [&](const float* src, unsigned short* dst, size_t n) {
    cast_kernel<<<(unsigned)((n + 255) / 256), 256, 0, stream>>>(src, dst, n);
  };
  cast(embedW, wEmbB, (size_t)DMODEL * SEQ);
  cast(Wq, wqB, (size_t)LAYERS * DMODEL * DMODEL);
  cast(Wk, wkB, (size_t)LAYERS * DMODEL * DMODEL);
  cast(Wv, wvB, (size_t)LAYERS * DMODEL * DMODEL);
  cast(Wo, woB, (size_t)LAYERS * DMODEL * DMODEL);
  cast(W3, w3B, (size_t)LAYERS * DMODEL * DMODEL);
  cast(W4, w4B, (size_t)LAYERS * DMODEL * DMODEL);
  cast(W1, w1B, (size_t)LAYERS * DFF * DMODEL);
  cast(W2, w2B, (size_t)LAYERS * DMODEL * DFF);
  for (int l = 0; l < LAYERS; ++l) {
    const size_t npad = (size_t)DBLKPAD * DFF;
    cast_pad_kernel<<<(unsigned)((npad + 255) / 256), 256, 0, stream>>>(
        W5 + (size_t)l * DBLK * DFF, w5B + (size_t)l * DBLKPAD * DFF, DBLK, DBLKPAD, DFF);
    cast_pad_kernel<<<(unsigned)((npad + 255) / 256), 256, 0, stream>>>(
        W6 + (size_t)l * DBLK * DFF, w6B + (size_t)l * DBLKPAD * DFF, DBLK, DBLKPAD, DFF);
  }

  stats_kernel<<<BT * ENC, 32, 0, stream>>>(x, meanB, stdB);
  build_xin_kernel<<<dim3(MPAD, BT), 128, 0, stream>>>(x, xmark, meanB, stdB, xinB);

  // h = xin @ embed_w^T + embed_b
  gemm_bf16_kernel<<<dim3(DMODEL / 64, MPAD / 32, BT), 32, 0, stream>>>(
      xinB, wEmbB, embedB, hF, hB, DMODEL, SEQ, 0, 0);

  const size_t nD  = nTok * DMODEL;
  const size_t nC  = nTok * DFF;
  const size_t nBl = nTok * DBLK;
  const dim3 gD(DMODEL / 64, MPAD / 32, BT);
  const dim3 gF(DFF / 64, MPAD / 32, BT);
  const dim3 gBlk(DBLKPAD / 64, MPAD / 32, BT);

  for (int l = 0; l < LAYERS; ++l) {
    const size_t wOff = (size_t)l * DMODEL * DMODEL;
    gemm_bf16_kernel<<<gD, 32, 0, stream>>>(hB, wqB + wOff, bq + l * DMODEL,
                                            (float*)nullptr, qB, DMODEL, DMODEL, 0, 0);
    gemm_bf16_kernel<<<gD, 32, 0, stream>>>(hB, wkB + wOff, bk + l * DMODEL,
                                            (float*)nullptr, kB, DMODEL, DMODEL, 0, 0);
    gemm_bf16_kernel<<<gD, 32, 0, stream>>>(hB, wvB + wOff, bv + l * DMODEL,
                                            (float*)nullptr, vtB, DMODEL, DMODEL, 0, 1);  // V^T

    attn_kernel<<<dim3(MPAD / 16, NHEAD, BT), 32, 0, stream>>>(qB, kB, vtB, attB);

    gemm_bf16_kernel<<<gD, 32, 0, stream>>>(attB, woB + wOff, bo + l * DMODEL,
                                            xattF, (unsigned short*)nullptr, DMODEL, DMODEL, 0, 0);

    // x_ln = LN(h - x_att)
    ln_kernel<<<(unsigned)nTok, 32, 0, stream>>>(hF, xattF, ln1g + l * DMODEL, ln1b + l * DMODEL, xlnF, xlnB);

    // y = GELU(x_ln W1^T + b1) W2^T + b2
    gemm_bf16_kernel<<<gF, 32, 0, stream>>>(xlnB, w1B + (size_t)l * DFF * DMODEL, b1 + l * DFF,
                                            (float*)nullptr, y1B, DFF, DMODEL, 1, 0);
    gemm_bf16_kernel<<<gD, 32, 0, stream>>>(y1B, w2B + (size_t)l * DMODEL * DFF, b2 + l * DMODEL,
                                            yF, (unsigned short*)nullptr, DMODEL, DFF, 0, 0);

    // cat = [x_att, y]
    concat_bf16_kernel<<<(unsigned)((nC + 255) / 256), 256, 0, stream>>>(xattF, yF, catB, nC);

    // xs = x_ln - y ; hh = sigmoid(xs W3^T + b3) * (xs W4^T + b4)
    sub_bf16_kernel<<<(unsigned)((nD + 255) / 256), 256, 0, stream>>>(xlnF, yF, xsB, nD);
    gemm_bf16_kernel<<<gD, 32, 0, stream>>>(xsB, w3B + wOff, b3 + l * DMODEL,
                                            g3F, (unsigned short*)nullptr, DMODEL, DMODEL, 0, 0);
    gemm_bf16_kernel<<<gD, 32, 0, stream>>>(xsB, w4B + wOff, b4 + l * DMODEL,
                                            g4F, (unsigned short*)nullptr, DMODEL, DMODEL, 0, 0);
    gate_mul_kernel<<<(unsigned)((nD + 255) / 256), 256, 0, stream>>>(g3F, g4F, hhF, nD);

    // h = LN(hh)
    ln_kernel<<<(unsigned)nTok, 32, 0, stream>>>(hhF, (const float*)nullptr,
                                                 ln2g + l * DMODEL, ln2b + l * DMODEL, hF, hB);

    // out = sigmoid(cat W5^T + b5) * (cat W6^T + b6); output = out - output
    gemm_bf16_kernel<<<gBlk, 32, 0, stream>>>(catB, w5B + (size_t)l * DBLKPAD * DFF, b5 + l * DBLK,
                                              g5F, (unsigned short*)nullptr, DBLK, DFF, 0, 0);
    gemm_bf16_kernel<<<gBlk, 32, 0, stream>>>(catB, w6B + (size_t)l * DBLKPAD * DFF, b6 + l * DBLK,
                                              g6F, (unsigned short*)nullptr, DBLK, DFF, 0, 0);
    gate_sub_kernel<<<(unsigned)((nBl + 255) / 256), 256, 0, stream>>>(g5F, g6F, outF, nBl, (l == 0) ? 1 : 0);
  }

  const size_t nOut = (size_t)B_SZ * SEQ * ENC;
  final_kernel<<<(unsigned)((nOut + 255) / 256), 256, 0, stream>>>(outF, meanB, stdB, (float*)d_out);
}
